// GAT_28741921144978
// MI455X (gfx1250) — matrix operand order repaired and checked by reading, not compiled, not run on hardware
//
#include <hip/hip_runtime.h>
#include <hip/hip_bf16.h>

#define IN_DIM    512
#define NEG_SLOPE 0.2f
#define OUT_DIM   40

typedef __attribute__((ext_vector_type(16))) __bf16 v16bf;
typedef __attribute__((ext_vector_type(8)))  float  v8f;

// ---------- helpers ----------
__device__ __forceinline__ unsigned fenc(float f) {
    unsigned u = __float_as_uint(f);
    return (u & 0x80000000u) ? ~u : (u | 0x80000000u);
}
__device__ __forceinline__ float fdec(unsigned k) {
    unsigned u = (k & 0x80000000u) ? (k ^ 0x80000000u) : ~k;
    return __uint_as_float(u);
}

// ---------- fused convert + A-fragment pack ----------
// A fragment (16-bit 16x32, per CDNA5 ISA): lane L (row = L&15):
//   elems 0..7  -> K = kbA + 0..7,  elems 8..15 -> K = kbA + 16..23,  kbA = (L>=16)*8
// Packed stream: p = ((tile16_m * KC + kc) * 32 + lane) * 16 + j  (32B contiguous per lane)
__global__ void k_pack_a(const float* __restrict__ X, __bf16* __restrict__ Ap,
                         int M, int K) {
    int p = blockIdx.x * blockDim.x + threadIdx.x;
    if (p >= M * K) return;
    int j    = p & 15;
    int lane = (p >> 4) & 31;
    int rest = p >> 9;
    int KC   = K >> 5;
    int kc   = rest % KC;
    int t16  = rest / KC;
    int row  = (t16 << 4) + (lane & 15);
    int kbA  = (lane >> 4) << 3;
    int k    = (kc << 5) + kbA + ((j < 8) ? j : (8 + j));  // j>=8: +16+(j-8)
    Ap[p] = (__bf16)X[(size_t)row * K + k];
}

// ---------- fused convert + B-fragment pack (zero-pad cols >= Nn) ----------
// B fragment (16-bit 32x16): lane L (col = L&15): elems 0..7 -> K = kbB + 0..7, elems 8..15 -> K = kbB + 16..23, kbB = (L>=16)*8
__global__ void k_pack_b(const float* __restrict__ W, __bf16* __restrict__ Bp,
                         int K, int Nn, int ntiles16) {
    int total = ntiles16 * (K >> 5) * 512;
    int p = blockIdx.x * blockDim.x + threadIdx.x;
    if (p >= total) return;
    int j    = p & 15;
    int lane = (p >> 4) & 31;
    int rest = p >> 9;
    int KC   = K >> 5;
    int kc   = rest % KC;
    int t16  = rest / KC;
    int col  = (t16 << 4) + (lane & 15);
    int kbB  = (lane >> 4) << 3;
    int k    = (kc << 5) + kbB + ((j < 8) ? j : (8 + j));
    Bp[p] = (col < Nn) ? (__bf16)W[(size_t)k * Nn + col] : (__bf16)0.f;
}

// ---------- WMMA GEMM on packed fragments: wave per 32x32 tile ----------
// per K-step: 4 fragment loads (8x global_load_b128) -> 4x v_wmma_f32_16x16x32_bf16
__global__ void k_wmma_gemm_packed(const __bf16* __restrict__ Ap,
                                   const __bf16* __restrict__ Bp,
                                   float* __restrict__ C,
                                   int M, int K, int Nn) {
    const int wave = threadIdx.x >> 5;
    const int lane = threadIdx.x & 31;
    const int KC = K >> 5;
    const int tiles_m2 = M >> 5;
    const int tiles_n2 = (Nn + 31) >> 5;
    int tile = blockIdx.x * (blockDim.x >> 5) + wave;
    if (tile >= tiles_m2 * tiles_n2) return;
    // n-major: waves of a block share the same packed-B lines
    const int tn2 = tile / tiles_m2;
    const int tm2 = tile - tn2 * tiles_m2;

    const size_t fstride = 512;  // elements per (tile16, kc) fragment block
    const __bf16* a0p = Ap + ((size_t)(2 * tm2)     * KC * 32 + lane) * 16;
    const __bf16* a1p = Ap + ((size_t)(2 * tm2 + 1) * KC * 32 + lane) * 16;
    const __bf16* b0p = Bp + ((size_t)(2 * tn2)     * KC * 32 + lane) * 16;
    const __bf16* b1p = Bp + ((size_t)(2 * tn2 + 1) * KC * 32 + lane) * 16;

    v8f acc00 = {}, acc01 = {}, acc10 = {}, acc11 = {};
#pragma unroll 2
    for (int kc = 0; kc < KC; ++kc) {
        v16bf a0 = *(const v16bf*)(a0p + (size_t)kc * fstride);
        v16bf a1 = *(const v16bf*)(a1p + (size_t)kc * fstride);
        v16bf b0 = *(const v16bf*)(b0p + (size_t)kc * fstride);
        v16bf b1 = *(const v16bf*)(b1p + (size_t)kc * fstride);
        acc00 = __builtin_amdgcn_wmma_f32_16x16x32_bf16(false, a0, false, b0, (short)0, acc00, false, false);
        acc01 = __builtin_amdgcn_wmma_f32_16x16x32_bf16(false, a0, false, b1, (short)0, acc01, false, false);
        acc10 = __builtin_amdgcn_wmma_f32_16x16x32_bf16(false, a1, false, b0, (short)0, acc10, false, false);
        acc11 = __builtin_amdgcn_wmma_f32_16x16x32_bf16(false, a1, false, b1, (short)0, acc11, false, false);
    }
    // D layout: lane col = lane&15, VGPR j -> row = 8*(lane>=16) + j
    const int l15 = lane & 15, hi = lane >> 4;
    const int col0 = (tn2 << 5) + l15;
    const int col1 = col0 + 16;
    const int m0 = (tm2 << 5) + (hi << 3);
    const int m1 = m0 + 16;
    if (col0 < Nn) {
#pragma unroll
        for (int j = 0; j < 8; ++j) {
            C[(size_t)(m0 + j) * Nn + col0] = acc00[j];
            C[(size_t)(m1 + j) * Nn + col0] = acc10[j];
        }
    }
    if (col1 < Nn) {
#pragma unroll
        for (int j = 0; j < 8; ++j) {
            C[(size_t)(m0 + j) * Nn + col1] = acc01[j];
            C[(size_t)(m1 + j) * Nn + col1] = acc11[j];
        }
    }
}

// ---------- per-(node,head) attention dot products ----------
__global__ void k_attn_dots(const float* __restrict__ g,
                            const float* __restrict__ att_src,
                            const float* __restrict__ att_dst,
                            float* __restrict__ asrc, float* __restrict__ adst,
                            int nN, int H, int C) {
    int i = blockIdx.x * blockDim.x + threadIdx.x;
    if (i >= nN * H) return;
    int n = i / H, h = i - n * H;
    const float* gp = g + (size_t)n * H * C + (size_t)h * C;
    const float* as = att_src + h * C;
    const float* ad = att_dst + h * C;
    float s = 0.f, d = 0.f;
    for (int c = 0; c < C; ++c) { float v = gp[c]; s += v * as[c]; d += v * ad[c]; }
    asrc[i] = s; adst[i] = d;
}

// ---------- fills ----------
__global__ void k_fill_f32(float* p, float v, int n) {
    int i = blockIdx.x * blockDim.x + threadIdx.x;
    if (i < n) p[i] = v;
}
__global__ void k_fill_u32(unsigned* p, unsigned v, int n) {
    int i = blockIdx.x * blockDim.x + threadIdx.x;
    if (i < n) p[i] = v;
}

// ---------- edge pass 1: segment max of leaky-relu logits ----------
__global__ void k_edge_max(const int* __restrict__ ei, int E, int nN, int H,
                           const float* __restrict__ asrc,
                           const float* __restrict__ adst,
                           unsigned* __restrict__ segmax) {
    int Etot = E + nN;
    int i = blockIdx.x * blockDim.x + threadIdx.x;
    if (i >= Etot * H) return;
    int e = i / H, h = i - e * H;
    int s, d;
    if (e < E) { s = ei[e]; d = ei[E + e]; } else { s = d = e - E; }
    float lg = asrc[s * H + h] + adst[d * H + h];
    lg = lg > 0.f ? lg : NEG_SLOPE * lg;
    atomicMax(&segmax[d * H + h], fenc(lg));
}

// ---------- edge pass 2: ex = exp(logit - max), denom += ex ----------
__global__ void k_edge_exp(const int* __restrict__ ei, int E, int nN, int H,
                           const float* __restrict__ asrc,
                           const float* __restrict__ adst,
                           const unsigned* __restrict__ segmax,
                           float* __restrict__ exbuf,
                           float* __restrict__ den) {
    int Etot = E + nN;
    int i = blockIdx.x * blockDim.x + threadIdx.x;
    if (i >= Etot * H) return;
    int e = i / H, h = i - e * H;
    int s, d;
    if (e < E) { s = ei[e]; d = ei[E + e]; } else { s = d = e - E; }
    float lg = asrc[s * H + h] + adst[d * H + h];
    lg = lg > 0.f ? lg : NEG_SLOPE * lg;
    float m = fdec(segmax[d * H + h]);
    if (!__builtin_isfinite(m)) m = 0.f;
    float ex = expf(lg - m);
    exbuf[i] = ex;
    atomicAdd(&den[d * H + h], ex);
}

// ---------- edge pass 3: out[dst] += alpha * g[src] (wave per edge) ----------
__global__ void k_edge_scatter(const int* __restrict__ ei, int E, int nN,
                               int H, int C,
                               const float* __restrict__ g,
                               const float* __restrict__ exbuf,
                               const float* __restrict__ den,
                               float* __restrict__ out) {
    int F = H * C;
    int Etot = E + nN;
    int wid = (int)((blockIdx.x * blockDim.x + threadIdx.x) >> 5);
    int lane = threadIdx.x & 31;
    if (wid >= Etot) return;
    int s, d;
    if (wid < E) { s = ei[wid]; d = ei[E + wid]; } else { s = d = wid - E; }
    for (int c = lane; c < F; c += 32) {
        int h = c / C;
        float alpha = exbuf[(size_t)wid * H + h] / (den[d * H + h] + 1e-16f);
        atomicAdd(&out[(size_t)d * F + c], g[(size_t)s * F + c] * alpha);
    }
}

// ---------- epilogue: bias + ELU (layers 1,2) ----------
__global__ void k_bias_elu(float* __restrict__ out, const float* __restrict__ bias,
                           int total, int F) {
    int i = blockIdx.x * blockDim.x + threadIdx.x;
    if (i >= total) return;
    float v = out[i] + bias[i % F];
    out[i] = v > 0.f ? v : (expf(v) - 1.f);
}

// ---------- final: bias + log_softmax ----------
__global__ void k_bias_logsoftmax(const float* __restrict__ out3,
                                  const float* __restrict__ bias,
                                  float* __restrict__ y, int nN, int F) {
    int n = blockIdx.x * blockDim.x + threadIdx.x;
    if (n >= nN) return;
    float buf[OUT_DIM];
    float mx = -1e30f;
    for (int f = 0; f < F; ++f) {
        float v = out3[(size_t)n * F + f] + bias[f];
        buf[f] = v;
        mx = fmaxf(mx, v);
    }
    float ssum = 0.f;
    for (int f = 0; f < F; ++f) ssum += expf(buf[f] - mx);
    float lse = mx + logf(ssum);
    for (int f = 0; f < F; ++f) y[(size_t)n * F + f] = buf[f] - lse;
}

extern "C" void kernel_launch(void* const* d_in, const int* in_sizes, int n_in,
                              void* d_out, int out_size, void* d_ws, size_t ws_size,
                              hipStream_t stream) {
    (void)n_in; (void)out_size; (void)ws_size;
    const float* x  = (const float*)d_in[0];
    const int*   ei = (const int*)d_in[1];
    const float* W[3]  = { (const float*)d_in[2], (const float*)d_in[6],  (const float*)d_in[10] };
    const float* aS[3] = { (const float*)d_in[3], (const float*)d_in[7],  (const float*)d_in[11] };
    const float* aD[3] = { (const float*)d_in[4], (const float*)d_in[8],  (const float*)d_in[12] };
    const float* bb[3] = { (const float*)d_in[5], (const float*)d_in[9],  (const float*)d_in[13] };

    const int nN   = in_sizes[0] / IN_DIM;   // 20000
    const int E    = in_sizes[1] / 2;        // 320000
    const int Etot = E + nN;

    // carve workspace
    char* ws = (char*)d_ws;
    size_t off = 0;
    auto carve = [&](size_t bytes) -> void* {
        void* p = ws + off;
        off = (off + bytes + 255) & ~(size_t)255;
        return p;
    };
    float*    g      = (float*)   carve((size_t)nN * 512 * sizeof(float));
    float*    out    = (float*)   carve((size_t)nN * 512 * sizeof(float));
    __bf16*   hbf    = (__bf16*)  carve((size_t)nN * 512 * sizeof(__bf16));    // packed A
    __bf16*   wbf    = (__bf16*)  carve((size_t)32 * 16 * 512 * sizeof(__bf16)); // packed B (max 32 col-tiles)
    float*    asrc   = (float*)   carve((size_t)nN * 8 * sizeof(float));
    float*    adst   = (float*)   carve((size_t)nN * 8 * sizeof(float));
    unsigned* segmax = (unsigned*)carve((size_t)nN * 8 * sizeof(unsigned));
    float*    den    = (float*)   carve((size_t)nN * 8 * sizeof(float));
    float*    exbuf  = (float*)   carve((size_t)Etot * 8 * sizeof(float));

    const unsigned NEG_INF_ENC = 0x007FFFFFu; // fenc(-inf)
    const int TB = 256;
    auto cdiv = [](long long a, long long b) { return (int)((a + b - 1) / b); };

    for (int layer = 0; layer < 3; ++layer) {
        const int H  = (layer < 2) ? 8 : 1;
        const int C  = (layer < 2) ? 64 : OUT_DIM;
        const int Nn = H * C;                    // 512, 512, 40
        const int K  = 512;
        const int tiles_n2  = (Nn + 31) >> 5;    // 16, 16, 2
        const int ntiles16  = tiles_n2 * 2;      // packed B col-tiles (zero-padded)
        const float* inp = (layer == 0) ? x : out;

        // 1) fused convert + fragment pack
        k_pack_a<<<cdiv((long long)nN * K, TB), TB, 0, stream>>>(inp, hbf, nN, K);
        k_pack_b<<<cdiv((long long)ntiles16 * (K >> 5) * 512, TB), TB, 0, stream>>>(
            W[layer], wbf, K, Nn, ntiles16);

        // 2) GEMM via WMMA (wave per 32x32 tile, 8 waves/block)
        {
            int tiles = (nN >> 5) * tiles_n2;
            k_wmma_gemm_packed<<<cdiv(tiles, TB / 32), TB, 0, stream>>>(hbf, wbf, g, nN, K, Nn);
        }

        // 3) attention dots
        k_attn_dots<<<cdiv((long long)nN * H, TB), TB, 0, stream>>>(
            g, aS[layer], aD[layer], asrc, adst, nN, H, C);

        // 4) init segment state + output accumulator
        k_fill_u32<<<cdiv((long long)nN * H, TB), TB, 0, stream>>>(segmax, NEG_INF_ENC, nN * H);
        k_fill_f32<<<cdiv((long long)nN * H, TB), TB, 0, stream>>>(den, 0.f, nN * H);
        k_fill_f32<<<cdiv((long long)nN * Nn, TB), TB, 0, stream>>>(out, 0.f, nN * Nn);

        // 5) three-pass segment softmax + scatter
        k_edge_max<<<cdiv((long long)Etot * H, TB), TB, 0, stream>>>(
            ei, E, nN, H, asrc, adst, segmax);
        k_edge_exp<<<cdiv((long long)Etot * H, TB), TB, 0, stream>>>(
            ei, E, nN, H, asrc, adst, segmax, exbuf, den);
        k_edge_scatter<<<cdiv((long long)Etot * 32, TB), TB, 0, stream>>>(
            ei, E, nN, H, C, g, exbuf, den, out);

        // 6) epilogue
        if (layer < 2) {
            k_bias_elu<<<cdiv((long long)nN * Nn, TB), TB, 0, stream>>>(
                out, bb[layer], nN * Nn, Nn);
        } else {
            k_bias_logsoftmax<<<cdiv(nN, TB), TB, 0, stream>>>(
                out, bb[layer], (float*)d_out, nN, OUT_DIM);
        }
    }
}
